// AttentionHead_85590108275316
// MI455X (gfx1250) — matrix operand order repaired and checked by reading, not compiled, not run on hardware
//
#include <hip/hip_runtime.h>
#include <hip/hip_bf16.h>

// Problem constants (match reference)
#define BB 8
#define DD 1024
#define NN 2048

typedef __attribute__((ext_vector_type(16))) __bf16 v16bf;
typedef __attribute__((ext_vector_type(8)))  float  v8f;

union FragB16 {
    v16bf v;
    uint4 q[2];
};

// Load a 16-element bf16 fragment as two contiguous 16B chunks.
// A fragment: p1 = p0 + 16 (other 8 K values belong to partner half-wave).
// B fragment (row-major over j): same k order as A, p1 = p0 + 16.
__device__ __forceinline__ v16bf load16(const __bf16* p0, const __bf16* p1) {
    FragB16 f;
    f.q[0] = *(const uint4*)p0;
    f.q[1] = *(const uint4*)p1;
    return f.v;
}

__device__ __forceinline__ v8f wmma_bf16(v16bf a, v16bf b, v8f c) {
    return __builtin_amdgcn_wmma_f32_16x16x32_bf16(
        /*neg_a=*/false, a, /*neg_b=*/false, b,
        /*c_mod=*/(short)0, c, /*reuse_a=*/false, /*reuse_b=*/false);
}

// ---------------------------------------------------------------- conversion
__global__ void cvt_f32_bf16(const float* __restrict__ src,
                             __bf16* __restrict__ dst, long n) {
    long i = (long)blockIdx.x * blockDim.x + threadIdx.x;
    long stride = (long)gridDim.x * blockDim.x;
    for (; i < n; i += stride) dst[i] = (__bf16)src[i];
}

// ------------------------------------------------- transpose + convert (LDS)
// x: [B][D][N] f32  ->  Xt: [B][N][D] bf16, 32x32 tiles via padded LDS.
__global__ void transpose_cvt(const float* __restrict__ x,
                              __bf16* __restrict__ Xt) {
    const int b  = blockIdx.z;
    const int e0 = blockIdx.y * 32;  // D index
    const int n0 = blockIdx.x * 32;  // N index
    const int tx = threadIdx.x;      // 0..31
    const int ty = threadIdx.y;      // 0..7
    __shared__ __bf16 tile[32][33];  // +1 pad to dodge bank conflicts
    const float* xb = x + ((size_t)b * DD + e0) * NN + n0;
#pragma unroll
    for (int j = 0; j < 4; ++j)
        tile[ty + 8 * j][tx] = (__bf16)xb[(size_t)(ty + 8 * j) * NN + tx];
    __syncthreads();
    __bf16* xt = Xt + ((size_t)b * NN + n0) * DD + e0;
#pragma unroll
    for (int j = 0; j < 4; ++j)
        xt[(size_t)(ty + 8 * j) * DD + tx] = tile[tx][ty + 8 * j];
}

// --------------------------------------------------------- generic NT GEMM
// C[b][i][j] = scale * sum_k A[b][i][k] * B[b][j][k]
// A: row-major M x K, B: row-major J x K, C: row-major M x J.
// One wave per 64x32 C block (4x2 grid of 16x16 tiles), explicit
// double-buffered fragments: 8 v_wmma per 32-wide K step, no spills.
__device__ __forceinline__ void load_frags(const __bf16* const* arow,
                                           const __bf16* const* brow, int k0,
                                           int klo, int kb, v16bf* a,
                                           v16bf* b) {
#pragma unroll
    for (int t = 0; t < 4; ++t)
        a[t] = load16(arow[t] + k0 + klo, arow[t] + k0 + 16 + klo);
#pragma unroll
    for (int t = 0; t < 2; ++t)
        b[t] = load16(brow[t] + k0 + klo, brow[t] + k0 + 16 + klo);
}

template <bool OUT_F32>
__global__ __launch_bounds__(32, 1) void gemm_nt(
    const __bf16* __restrict__ A, const __bf16* __restrict__ B,
    void* __restrict__ C, int J, int K, long strideA, long strideB,
    long strideC, float scale) {
    const int lane = threadIdx.x;  // 0..31
    const int b  = blockIdx.z;
    const int i0 = blockIdx.y * 64;
    const int j0 = blockIdx.x * 32;
    const __bf16* Ab = A + (size_t)b * strideA;
    const __bf16* Bb = B + (size_t)b * strideB;

    const int klo = (lane & 16) ? 8 : 0;  // A-frag K half select
    const int kb  = lane & 16;            // unused; B-frag also uses klo
    const __bf16* arow[4];
    const __bf16* brow[2];
#pragma unroll
    for (int t = 0; t < 4; ++t)
        arow[t] = Ab + (size_t)(i0 + 16 * t + (lane & 15)) * K;
#pragma unroll
    for (int t = 0; t < 2; ++t)
        brow[t] = Bb + (size_t)(j0 + 16 * t + (lane & 15)) * K;

    v8f acc[4][2] = {};
    v16bf aX[4], bX[2], aY[4], bY[2];
    load_frags(arow, brow, 0, klo, kb, aX, bX);

    for (int k0 = 0; k0 < K; k0 += 64) {
        // prefetch far-ahead chunk into L2 while we compute
        if (k0 + 128 < K) {
            __builtin_prefetch(arow[0] + k0 + 128, 0, 1);
            __builtin_prefetch(brow[0] + k0 + 128, 0, 1);
        }
        load_frags(arow, brow, k0 + 32, klo, kb, aY, bY);
#pragma unroll
        for (int i = 0; i < 4; ++i)
#pragma unroll
            for (int j = 0; j < 2; ++j)
                acc[i][j] = wmma_bf16(aX[i], bX[j], acc[i][j]);
        if (k0 + 64 < K) load_frags(arow, brow, k0 + 64, klo, kb, aX, bX);
#pragma unroll
        for (int i = 0; i < 4; ++i)
#pragma unroll
            for (int j = 0; j < 2; ++j)
                acc[i][j] = wmma_bf16(aY[i], bY[j], acc[i][j]);
    }

    const int col = lane & 15;
    const int rbase = (lane & 16) ? 8 : 0;  // C rows 0-7 (lo) / 8-15 (hi)
#pragma unroll
    for (int i = 0; i < 4; ++i) {
#pragma unroll
        for (int r = 0; r < 8; ++r) {
            const size_t row = (size_t)(i0 + 16 * i + rbase + r);
#pragma unroll
            for (int j = 0; j < 2; ++j) {
                const size_t idx =
                    (size_t)b * strideC + row * J + (j0 + 16 * j + col);
                const float v = acc[i][j][r] * scale;
                if (OUT_F32) ((float*)C)[idx] = v;
                else         ((__bf16*)C)[idx] = (__bf16)v;
            }
        }
    }
}

// ------------------------------------------------------------------- softmax
// St: [B][m=NN][n=NN]; softmax over m (rows) for each column n.
// Each thread owns one column -> perfectly coalesced, no reductions needed.
__global__ void softmax_cols(__bf16* __restrict__ St) {
    const int n = blockIdx.x * blockDim.x + threadIdx.x;
    const int b = blockIdx.y;
    __bf16* p = St + (size_t)b * NN * NN + n;
    float mx = -3.0e38f;
    for (int m = 0; m < NN; ++m) mx = fmaxf(mx, (float)p[(size_t)m * NN]);
    float s = 0.0f;
    for (int m = 0; m < NN; ++m) s += __expf((float)p[(size_t)m * NN] - mx);
    const float inv = 1.0f / s;
    for (int m = 0; m < NN; ++m) {
        const size_t idx = (size_t)m * NN;
        p[idx] = (__bf16)(__expf((float)p[idx] - mx) * inv);
    }
}

// ------------------------------------------------------------------ launcher
extern "C" void kernel_launch(void* const* d_in, const int* in_sizes, int n_in,
                              void* d_out, int out_size, void* d_ws,
                              size_t ws_size, hipStream_t stream) {
    const float* x  = (const float*)d_in[0];
    const float* Wq = (const float*)d_in[1];
    const float* Wk = (const float*)d_in[2];
    const float* Wv = (const float*)d_in[3];
    float* out = (float*)d_out;

    char* ws = (char*)d_ws;
    size_t off = 0;
    auto alloc = [&](size_t bytes) {
        char* p = ws + off;
        off += (bytes + 255) & ~(size_t)255;
        return p;
    };
    __bf16* Wqb = (__bf16*)alloc((size_t)DD * DD * 2);
    __bf16* Wkb = (__bf16*)alloc((size_t)DD * DD * 2);
    __bf16* Wvb = (__bf16*)alloc((size_t)DD * DD * 2);
    __bf16* Xt  = (__bf16*)alloc((size_t)BB * NN * DD * 2);  // x transposed
    __bf16* Qt  = (__bf16*)alloc((size_t)BB * NN * DD * 2);  // [B,N,D], scaled
    __bf16* Kt  = (__bf16*)alloc((size_t)BB * NN * DD * 2);  // [B,N,D]
    __bf16* Vb  = (__bf16*)alloc((size_t)BB * DD * NN * 2);  // [B,D,N]
    __bf16* St  = (__bf16*)alloc((size_t)BB * NN * NN * 2);  // scores^T -> P^T

    // bf16 weights; x transposed+converted through LDS tiles
    cvt_f32_bf16<<<256, 256, 0, stream>>>(Wq, Wqb, (long)DD * DD);
    cvt_f32_bf16<<<256, 256, 0, stream>>>(Wk, Wkb, (long)DD * DD);
    cvt_f32_bf16<<<256, 256, 0, stream>>>(Wv, Wvb, (long)DD * DD);
    transpose_cvt<<<dim3(NN / 32, DD / 32, BB), dim3(32, 8), 0, stream>>>(x, Xt);

    const float inv_sqrt_d = 0.03125f;  // 1/sqrt(1024)
    const long sXD = (long)NN * DD;     // per-batch stride of [N,D]/[D,N]
    const long sNN = (long)NN * NN;
    dim3 wave(32);

    // Qt[n][d] = sum_e Xt[n][e] * Wq[d][e]   (scaled)
    gemm_nt<false><<<dim3(DD / 32, NN / 64, BB), wave, 0, stream>>>(
        Xt, Wqb, Qt, DD, DD, sXD, 0, sXD, inv_sqrt_d);
    // Kt[n][d] = sum_e Xt[n][e] * Wk[d][e]
    gemm_nt<false><<<dim3(DD / 32, NN / 64, BB), wave, 0, stream>>>(
        Xt, Wkb, Kt, DD, DD, sXD, 0, sXD, 1.0f);
    // V[d][n] = sum_e Wv[d][e] * Xt[n][e]
    gemm_nt<false><<<dim3(NN / 32, DD / 64, BB), wave, 0, stream>>>(
        Wvb, Xt, Vb, NN, DD, 0, sXD, sXD, 1.0f);

    // St[m][n] = sum_d Kt[m][d] * Qt[n][d]  (= scores[b,n,m])
    gemm_nt<false><<<dim3(NN / 32, NN / 64, BB), wave, 0, stream>>>(
        Kt, Qt, St, NN, DD, sXD, sXD, sNN, 1.0f);

    // softmax over keys m (rows of St) for each query column n
    softmax_cols<<<dim3(NN / 256, BB), 256, 0, stream>>>(St);

    // out[b][d][m] = sum_n V[d][n] * Pt[m][n]  (f32 output)
    gemm_nt<true><<<dim3(NN / 32, DD / 64, BB), wave, 0, stream>>>(
        Vb, St, out, NN, NN, sXD, sNN, (long)DD * NN, 1.0f);
}
